// MP_PDE_Solver_46488726012232
// MI455X (gfx1250) — matrix operand order repaired and checked by reading, not compiled, not run on hardware
//
#include <hip/hip_runtime.h>

// ---------------------------------------------------------------------------
// MP-PDE GNN for MI455X (gfx1250, wave32).
// Keep h in fp16 (8MB -> L2 resident), fuse both message-MLP layers per
// 128-edge tile with v_wmma_f32_16x16x32_f16, scatter segment-sum with global
// f32 atomics into an L2-resident accumulator. Each wave owns TWO 16-row tiles
// so every weight (B) fragment loaded from LDS feeds two WMMAs -> halves ds
// traffic per edge vs one-tile-per-wave.
// ---------------------------------------------------------------------------

typedef __attribute__((ext_vector_type(16))) _Float16 v16h;
typedef __attribute__((ext_vector_type(8)))  float    v8f;

#define DEV __device__ __forceinline__

static const int N_NODES = 65536;
static const int N_EDGES = 1 << 20;
#define MSG_KPAD 160   // 133 padded to 5 chunks of 32
#define UPD_K    128

DEV float swishf(float x) { return x / (1.0f + __expf(-x)); }

DEV v8f zero8() { v8f z = {0.f,0.f,0.f,0.f,0.f,0.f,0.f,0.f}; return z; }

DEV v8f wmma16(v16h a, v16h b, v8f c) {
    return __builtin_amdgcn_wmma_f32_16x16x32_f16(
        /*neg_a=*/false, a, /*neg_b=*/false, b,
        /*c_mod=*/(short)0, c, /*reuse_a=*/false, /*reuse_b=*/false);
}

// A fragment: 16 rows x 32 K (f16). ISA layout: lanes 0-15 = rows, halves
// 0..7 -> K = base+0..7, halves 8..15 -> K = base+16..23; lanes>=16 add +8 to K.
DEV v16h load_a_frag(const _Float16* base, int strideH, int kBase, int lane) {
    const _Float16* p = base + (lane & 15) * strideH + kBase + ((lane >> 4) << 3);
    v16h a;
#pragma unroll
    for (int i = 0; i < 8; i++) { a[i] = p[i]; a[8 + i] = p[16 + i]; }
    return a;
}

// B fragment: 32 K x 16 N (f16), staged transposed in LDS as wT[n][k].
// lanes 0-15 = columns, halves 0..7 -> K=kBase+0..7, halves 8..15 -> K=kBase+16..23; lanes>=16 add +8 to K.
DEV v16h load_b_frag(const _Float16* wT, int strideH, int kBase, int lane, int nBase) {
    const _Float16* p = wT + (nBase + (lane & 15)) * strideH + kBase + ((lane >> 4) << 3);
    v16h b;
#pragma unroll
    for (int i = 0; i < 8; i++) { b[i] = p[i]; b[8 + i] = p[16 + i]; }
    return b;
}

// ---------------------------------------------------------------------------
// Fused edge message kernel: gather -> MLP1(133->64) -> MLP2(64->64) -> atomic
// scatter-add into agg. Block = 128 threads (4 waves), 128 edges per block;
// each wave computes 32 edges as two 16-row tiles sharing B fragments.
// ---------------------------------------------------------------------------
__global__ __launch_bounds__(128)
void msg_kernel(const _Float16* __restrict__ hg,     // N x 64 f16
                const _Float16* __restrict__ ef,     // E x 8 f16 (du0..2,dp0..1,0,0,0)
                const int* __restrict__ srcIdx,
                const int* __restrict__ dstIdx,
                const _Float16* __restrict__ w1T,    // 64 x 160 f16 (transposed, padded)
                const float* __restrict__ b1,
                const _Float16* __restrict__ w2T,    // 64 x 64 f16 (transposed)
                const float* __restrict__ b2,
                float* __restrict__ agg)             // N x 64 f32
{
    __shared__ __align__(16) _Float16 sIn[128 * MSG_KPAD];   // 40 KB
    __shared__ __align__(16) _Float16 sW1T[64 * MSG_KPAD];   // 20 KB
    __shared__ __align__(16) _Float16 sW2T[64 * 64];         //  8 KB
    __shared__ __align__(16) _Float16 sHid[4][32 * 64];      // 16 KB

    const int tid = threadIdx.x, lane = tid & 31, wave = tid >> 5;
    const int eBase = blockIdx.x * 128;

    // stage weights (vectorized)
    for (int i = tid; i < 64 * MSG_KPAD / 8; i += 128)
        ((float4*)sW1T)[i] = ((const float4*)w1T)[i];
    for (int i = tid; i < 64 * 64 / 8; i += 128)
        ((float4*)sW2T)[i] = ((const float4*)w2T)[i];

    // stage gathered m_in rows: [h_dst(64) | h_src(64) | du,dpos(5) | pad->160]
    for (int i = tid; i < 128 * 16; i += 128) {
        int e = i >> 4, c = i & 15;
        int eg = eBase + e;
        float4 v;
        if (c < 8) { int d = dstIdx[eg]; v = ((const float4*)(hg + (size_t)d * 64))[c]; }
        else       { int s = srcIdx[eg]; v = ((const float4*)(hg + (size_t)s * 64))[c - 8]; }
        ((float4*)(sIn + e * MSG_KPAD))[c] = v;
    }
    for (int i = tid; i < 128 * 4; i += 128) {
        int e = i >> 2, c = i & 3;
        float4 z = {0.f, 0.f, 0.f, 0.f};
        if (c == 0) z = ((const float4*)(ef + (size_t)(eBase + e) * 8))[0];
        ((float4*)(sIn + e * MSG_KPAD + 128))[c] = z;
    }
    __syncthreads();

    // GEMM1: (2 x 16 edges x 160) @ (160 x 64) per wave, B fragment reused x2
    v8f acc[2][4];
#pragma unroll
    for (int t = 0; t < 2; t++)
#pragma unroll
        for (int ct = 0; ct < 4; ct++) acc[t][ct] = zero8();

    const _Float16* aBase = sIn + wave * 32 * MSG_KPAD;
#pragma unroll
    for (int kc = 0; kc < 5; kc++) {
        v16h a0 = load_a_frag(aBase,                  MSG_KPAD, kc * 32, lane);
        v16h a1 = load_a_frag(aBase + 16 * MSG_KPAD,  MSG_KPAD, kc * 32, lane);
#pragma unroll
        for (int ct = 0; ct < 4; ct++) {
            v16h b = load_b_frag(sW1T, MSG_KPAD, kc * 32, lane, ct * 16);
            acc[0][ct] = wmma16(a0, b, acc[0][ct]);
            acc[1][ct] = wmma16(a1, b, acc[1][ct]);
        }
    }

    const int n0 = lane & 15, mh = (lane >> 4) << 3;
#pragma unroll
    for (int t = 0; t < 2; t++)
#pragma unroll
        for (int ct = 0; ct < 4; ct++) {
            int n = ct * 16 + n0;
            float bb = b1[n];
#pragma unroll
            for (int r = 0; r < 8; r++)
                sHid[wave][(t * 16 + mh + r) * 64 + n] = (_Float16)swishf(acc[t][ct][r] + bb);
        }
    __syncthreads();

    // GEMM2: (2 x 16 x 64) @ (64 x 64)
    v8f acc2[2][4];
#pragma unroll
    for (int t = 0; t < 2; t++)
#pragma unroll
        for (int ct = 0; ct < 4; ct++) acc2[t][ct] = zero8();

#pragma unroll
    for (int kc = 0; kc < 2; kc++) {
        v16h a0 = load_a_frag(sHid[wave],           64, kc * 32, lane);
        v16h a1 = load_a_frag(sHid[wave] + 16 * 64, 64, kc * 32, lane);
#pragma unroll
        for (int ct = 0; ct < 4; ct++) {
            v16h b = load_b_frag(sW2T, 64, kc * 32, lane, ct * 16);
            acc2[0][ct] = wmma16(a0, b, acc2[0][ct]);
            acc2[1][ct] = wmma16(a1, b, acc2[1][ct]);
        }
    }

    // swish + segment-sum via f32 atomics (agg is L2 resident)
#pragma unroll
    for (int t = 0; t < 2; t++)
#pragma unroll
        for (int ct = 0; ct < 4; ct++) {
            int n = ct * 16 + n0;
            float bb = b2[n];
#pragma unroll
            for (int r = 0; r < 8; r++) {
                float v = swishf(acc2[t][ct][r] + bb);
                int e = eBase + wave * 32 + t * 16 + mh + r;
                int d = dstIdx[e];
                atomicAdd(&agg[(size_t)d * 64 + n], v);
            }
        }
}

// ---------------------------------------------------------------------------
// Fused node update kernel: [h | agg/deg] -> MLP(128->64) -> MLP(64->64),
// residual add, per-(graph,channel) instance-norm stats via atomics.
// Block = 128 threads (4 waves), 128 nodes per block (4096 nodes per graph,
// blocks aligned -> all nodes in one block share a graph).
// ---------------------------------------------------------------------------
__global__ __launch_bounds__(128)
void upd_kernel(const _Float16* __restrict__ hg, const float* __restrict__ h32,
                const float* __restrict__ agg, const float* __restrict__ invdeg,
                const _Float16* __restrict__ w1T, const float* __restrict__ b1,
                const _Float16* __restrict__ w2T, const float* __restrict__ b2,
                float* __restrict__ hpre, float* __restrict__ stats)
{
    __shared__ __align__(16) _Float16 sIn[128 * UPD_K];      // 32 KB
    __shared__ __align__(16) _Float16 sW1T[64 * UPD_K];      // 16 KB
    __shared__ __align__(16) _Float16 sW2T[64 * 64];         //  8 KB
    __shared__ __align__(16) _Float16 sHid[4][32 * 64];      // 16 KB

    const int tid = threadIdx.x, lane = tid & 31, wave = tid >> 5;
    const int nodeBase = blockIdx.x * 128;

    for (int i = tid; i < 64 * UPD_K / 8; i += 128)
        ((float4*)sW1T)[i] = ((const float4*)w1T)[i];
    for (int i = tid; i < 64 * 64 / 8; i += 128)
        ((float4*)sW2T)[i] = ((const float4*)w2T)[i];

    for (int i = tid; i < 128 * 8; i += 128) {       // h part (contiguous, vectorized)
        int r = i >> 3, c = i & 7;
        ((float4*)(sIn + r * UPD_K))[c] = ((const float4*)(hg + (size_t)(nodeBase + r) * 64))[c];
    }
    for (int i = tid; i < 128 * 64; i += 128) {      // agg/deg part
        int r = i >> 6, c = i & 63;
        int node = nodeBase + r;
        sIn[r * UPD_K + 64 + c] = (_Float16)(agg[(size_t)node * 64 + c] * invdeg[node]);
    }
    __syncthreads();

    v8f acc[2][4];
#pragma unroll
    for (int t = 0; t < 2; t++)
#pragma unroll
        for (int ct = 0; ct < 4; ct++) acc[t][ct] = zero8();

    const _Float16* aBase = sIn + wave * 32 * UPD_K;
#pragma unroll
    for (int kc = 0; kc < 4; kc++) {
        v16h a0 = load_a_frag(aBase,               UPD_K, kc * 32, lane);
        v16h a1 = load_a_frag(aBase + 16 * UPD_K,  UPD_K, kc * 32, lane);
#pragma unroll
        for (int ct = 0; ct < 4; ct++) {
            v16h b = load_b_frag(sW1T, UPD_K, kc * 32, lane, ct * 16);
            acc[0][ct] = wmma16(a0, b, acc[0][ct]);
            acc[1][ct] = wmma16(a1, b, acc[1][ct]);
        }
    }

    const int n0 = lane & 15, mh = (lane >> 4) << 3;
#pragma unroll
    for (int t = 0; t < 2; t++)
#pragma unroll
        for (int ct = 0; ct < 4; ct++) {
            int n = ct * 16 + n0;
            float bb = b1[n];
#pragma unroll
            for (int r = 0; r < 8; r++)
                sHid[wave][(t * 16 + mh + r) * 64 + n] = (_Float16)swishf(acc[t][ct][r] + bb);
        }
    __syncthreads();

    v8f acc2[2][4];
#pragma unroll
    for (int t = 0; t < 2; t++)
#pragma unroll
        for (int ct = 0; ct < 4; ct++) acc2[t][ct] = zero8();

#pragma unroll
    for (int kc = 0; kc < 2; kc++) {
        v16h a0 = load_a_frag(sHid[wave],           64, kc * 32, lane);
        v16h a1 = load_a_frag(sHid[wave] + 16 * 64, 64, kc * 32, lane);
#pragma unroll
        for (int ct = 0; ct < 4; ct++) {
            v16h b = load_b_frag(sW2T, 64, kc * 32, lane, ct * 16);
            acc2[0][ct] = wmma16(a0, b, acc2[0][ct]);
            acc2[1][ct] = wmma16(a1, b, acc2[1][ct]);
        }
    }

    const int graph = nodeBase >> 12;   // 4096 nodes per graph, contiguous
#pragma unroll
    for (int t = 0; t < 2; t++)
#pragma unroll
        for (int ct = 0; ct < 4; ct++) {
            int n = ct * 16 + n0;
            float bb = b2[n];
#pragma unroll
            for (int r = 0; r < 8; r++) {
                int node = nodeBase + wave * 32 + t * 16 + mh + r;
                float v = h32[(size_t)node * 64 + n] + swishf(acc2[t][ct][r] + bb);
                hpre[(size_t)node * 64 + n] = v;
                atomicAdd(&stats[graph * 128 + n], v);
                atomicAdd(&stats[graph * 128 + 64 + n], v * v);
            }
        }
}

// ---------------------------------------------------------------------------
// Small helper kernels
// ---------------------------------------------------------------------------
__global__ void zero_kernel(float* p, int n) {
    int i = blockIdx.x * 256 + threadIdx.x;
    if (i < n) p[i] = 0.0f;
}

__global__ void deg_kernel(const int* __restrict__ dst, float* __restrict__ deg) {
    int e = blockIdx.x * 256 + threadIdx.x;
    if (e < N_EDGES) atomicAdd(&deg[dst[e]], 1.0f);
}

__global__ void invdeg_kernel(float* deg) {
    int i = blockIdx.x * 256 + threadIdx.x;
    if (i < N_NODES) deg[i] = 1.0f / fmaxf(deg[i], 1.0f);
}

// inputs row = [pos0,pos1,u0,u1,u2]; edge feat = [du(3), dpos(2), 0,0,0] f16
__global__ void edgefeat_kernel(const float* __restrict__ in, const int* __restrict__ src,
                                const int* __restrict__ dst, _Float16* __restrict__ ef) {
    int e = blockIdx.x * 256 + threadIdx.x;
    if (e >= N_EDGES) return;
    const float* rs = in + (size_t)src[e] * 5;
    const float* rd = in + (size_t)dst[e] * 5;
    _Float16* o = ef + (size_t)e * 8;
    o[0] = (_Float16)(rd[2] - rs[2]);
    o[1] = (_Float16)(rd[3] - rs[3]);
    o[2] = (_Float16)(rd[4] - rs[4]);
    o[3] = (_Float16)(rd[0] - rs[0]);
    o[4] = (_Float16)(rd[1] - rs[1]);
    o[5] = (_Float16)0.0f; o[6] = (_Float16)0.0f; o[7] = (_Float16)0.0f;
}

// W (K x 64, row-major f32) -> wT (64 x Kpad, f16, zero padded K rows)
__global__ void cvtw_kernel(const float* __restrict__ W, _Float16* __restrict__ wT,
                            int K, int Kpad) {
    int i = blockIdx.x * 256 + threadIdx.x;
    if (i >= 64 * Kpad) return;
    int n = i / Kpad, k = i % Kpad;
    wT[i] = (k < K) ? (_Float16)W[k * 64 + n] : (_Float16)0.0f;
}

// node embedding: 5 -> 64 -> 64, swish; writes h32 and h16
__global__ __launch_bounds__(64)
void emb_kernel(const float* __restrict__ in, const float* __restrict__ W1,
                const float* __restrict__ b1, const float* __restrict__ W2,
                const float* __restrict__ b2,
                float* __restrict__ h32, _Float16* __restrict__ h16) {
    __shared__ float sIn[5];
    __shared__ float sH[64];
    int node = blockIdx.x, o = threadIdx.x;
    if (o < 5) {
        int col = (o < 3) ? (2 + o) : (o - 3);   // node_in = [u(3), pos(2)]
        sIn[o] = in[(size_t)node * 5 + col];
    }
    __syncthreads();
    float a = b1[o];
    for (int k = 0; k < 5; k++) a += sIn[k] * W1[k * 64 + o];
    sH[o] = swishf(a);
    __syncthreads();
    float a2 = b2[o];
    for (int k = 0; k < 64; k++) a2 += sH[k] * W2[k * 64 + o];
    float h = swishf(a2);
    h32[(size_t)node * 64 + o] = h;
    h16[(size_t)node * 64 + o] = (_Float16)h;
}

__global__ void statsfin_kernel(const float* __restrict__ stats, float* __restrict__ ms) {
    int i = blockIdx.x * 256 + threadIdx.x;
    if (i >= 16 * 64) return;
    int g = i >> 6, n = i & 63;
    float s = stats[g * 128 + n], ss = stats[g * 128 + 64 + n];
    float mean = s * (1.0f / 4096.0f);
    float var = ss * (1.0f / 4096.0f) - mean * mean;
    ms[g * 128 + n] = mean;
    ms[g * 128 + 64 + n] = rsqrtf(var + 1e-5f);
}

__global__ void norm_kernel(const float* __restrict__ hpre, const float* __restrict__ ms,
                            float* __restrict__ h32, _Float16* __restrict__ h16) {
    int i = blockIdx.x * 256 + threadIdx.x;
    if (i >= N_NODES * 64) return;
    int node = i >> 6, n = i & 63, g = node >> 12;
    float v = (hpre[i] - ms[g * 128 + n]) * ms[g * 128 + 64 + n];
    h32[i] = v;
    h16[i] = (_Float16)v;
}

// output head: 64 -> 64 swish -> 3, out = u + diff
__global__ __launch_bounds__(64)
void out_kernel(const float* __restrict__ h32, const float* __restrict__ in,
                const float* __restrict__ W1, const float* __restrict__ b1,
                const float* __restrict__ W2, const float* __restrict__ b2,
                float* __restrict__ out) {
    __shared__ float sO[64];
    int node = blockIdx.x, j = threadIdx.x;
    const float* hr = h32 + (size_t)node * 64;
    float a = b1[j];
    for (int k = 0; k < 64; k++) a += hr[k] * W1[k * 64 + j];
    sO[j] = swishf(a);
    __syncthreads();
    if (j < 3) {
        float d = b2[j];
        for (int k = 0; k < 64; k++) d += sO[k] * W2[k * 3 + j];
        out[(size_t)node * 3 + j] = in[(size_t)node * 5 + 2 + j] + d;
    }
}

// ---------------------------------------------------------------------------
static inline int div_up(int a, int b) { return (a + b - 1) / b; }

extern "C" void kernel_launch(void* const* d_in, const int* in_sizes, int n_in,
                              void* d_out, int out_size, void* d_ws, size_t ws_size,
                              hipStream_t stream) {
    (void)in_sizes; (void)n_in; (void)out_size; (void)ws_size;
    const int N = N_NODES, E = N_EDGES;

    const float* inputs = (const float*)d_in[0];
    const int* src = (const int*)d_in[1];
    const int* dst = src + E;
    // d_in[2] = batch (contiguous: graph = node >> 12, unused)
    const float* embW1 = (const float*)d_in[3];
    const float* embB1 = (const float*)d_in[4];
    const float* embW2 = (const float*)d_in[5];
    const float* embB2 = (const float*)d_in[6];
    const float *mW1[6], *mb1[6], *mW2[6], *mb2[6], *uW1[6], *ub1[6], *uW2[6], *ub2[6];
    for (int l = 0; l < 6; l++) {
        int b = 7 + 8 * l;
        mW1[l] = (const float*)d_in[b + 0]; mb1[l] = (const float*)d_in[b + 1];
        mW2[l] = (const float*)d_in[b + 2]; mb2[l] = (const float*)d_in[b + 3];
        uW1[l] = (const float*)d_in[b + 4]; ub1[l] = (const float*)d_in[b + 5];
        uW2[l] = (const float*)d_in[b + 6]; ub2[l] = (const float*)d_in[b + 7];
    }
    const float* oW1 = (const float*)d_in[55];
    const float* ob1 = (const float*)d_in[56];
    const float* oW2 = (const float*)d_in[57];
    const float* ob2 = (const float*)d_in[58];

    // workspace carving (256B aligned)
    char* p = (char*)d_ws;
    auto alloc = [&](size_t bytes) { char* r = p; p += (bytes + 255) & ~(size_t)255; return r; };
    float*     h32    = (float*)alloc((size_t)N * 64 * 4);
    _Float16*  h16    = (_Float16*)alloc((size_t)N * 64 * 2);
    float*     hpre   = (float*)alloc((size_t)N * 64 * 4);
    float*     agg    = (float*)alloc((size_t)N * 64 * 4);
    float*     invdeg = (float*)alloc((size_t)N * 4);
    _Float16*  ef16   = (_Float16*)alloc((size_t)E * 8 * 2);
    float*     stats  = (float*)alloc(16 * 128 * 4);
    float*     ms     = (float*)alloc(16 * 128 * 4);
    _Float16 *mW1T[6], *mW2T[6], *uW1T[6], *uW2T[6];
    for (int l = 0; l < 6; l++) {
        mW1T[l] = (_Float16*)alloc(64 * MSG_KPAD * 2);
        mW2T[l] = (_Float16*)alloc(64 * 64 * 2);
        uW1T[l] = (_Float16*)alloc(64 * UPD_K * 2);
        uW2T[l] = (_Float16*)alloc(64 * 64 * 2);
    }

    // degrees -> inverse clipped degree
    zero_kernel<<<div_up(N, 256), 256, 0, stream>>>(invdeg, N);
    deg_kernel<<<div_up(E, 256), 256, 0, stream>>>(dst, invdeg);
    invdeg_kernel<<<div_up(N, 256), 256, 0, stream>>>(invdeg);

    // edge geometric features (precomputed once, f16)
    edgefeat_kernel<<<div_up(E, 256), 256, 0, stream>>>(inputs, src, dst, ef16);

    // weights -> transposed, padded f16
    for (int l = 0; l < 6; l++) {
        cvtw_kernel<<<div_up(64 * MSG_KPAD, 256), 256, 0, stream>>>(mW1[l], mW1T[l], 133, MSG_KPAD);
        cvtw_kernel<<<div_up(64 * 64, 256), 256, 0, stream>>>(mW2[l], mW2T[l], 64, 64);
        cvtw_kernel<<<div_up(64 * UPD_K, 256), 256, 0, stream>>>(uW1[l], uW1T[l], 128, UPD_K);
        cvtw_kernel<<<div_up(64 * 64, 256), 256, 0, stream>>>(uW2[l], uW2T[l], 64, 64);
    }

    // embedding
    emb_kernel<<<N, 64, 0, stream>>>(inputs, embW1, embB1, embW2, embB2, h32, h16);

    // message-passing layers
    for (int l = 0; l < 6; l++) {
        zero_kernel<<<div_up(N * 64, 256), 256, 0, stream>>>(agg, N * 64);
        zero_kernel<<<div_up(16 * 128, 256), 256, 0, stream>>>(stats, 16 * 128);
        msg_kernel<<<E / 128, 128, 0, stream>>>(h16, ef16, src, dst,
                                                mW1T[l], mb1[l], mW2T[l], mb2[l], agg);
        upd_kernel<<<N / 128, 128, 0, stream>>>(h16, h32, agg, invdeg,
                                                uW1T[l], ub1[l], uW2T[l], ub2[l], hpre, stats);
        statsfin_kernel<<<div_up(16 * 64, 256), 256, 0, stream>>>(stats, ms);
        norm_kernel<<<div_up(N * 64, 256), 256, 0, stream>>>(hpre, ms, h32, h16);
    }

    // output head
    out_kernel<<<N, 64, 0, stream>>>(h32, inputs, oW1, ob1, oW2, ob2, (float*)d_out);
}
